// TAMCaD_T_11776800325804
// MI455X (gfx1250) — matrix operand order repaired and checked by reading, not compiled, not run on hardware
//
#include <hip/hip_runtime.h>
#include <hip/hip_bf16.h>

// ---------------------------------------------------------------------------
// TAMCaD_T attention, MI455X (gfx1250, wave32, WMMA).
// Per (b,s): logits = scale * Q_s K_s^T ; attn = softmax(logits) ; X = attn V_s
// Q_s,K_s,V_s are 32x32.  One wave per s, one workgroup per (b, 32-s block).
// Memory-bound: all global traffic is coalesced along the innermost s axis;
// matmuls use v_wmma_f32_16x16x32_f16 (K=32 covers the full reduction).
// ---------------------------------------------------------------------------

typedef _Float16 v16h __attribute__((ext_vector_type(16)));
typedef _Float16 h8   __attribute__((ext_vector_type(8)));
typedef float    v8f  __attribute__((ext_vector_type(8)));

#define S_TOTAL 4096
#define GDIM    32           // groups (g / f)
#define DDIM    32           // hidden (d), n_heads = 1
#define SB      32           // s values per workgroup == waves per block

// f16 tile: [s][row][col], row stride 40 halves (80B, 16B aligned),
// s-slice stride 32*40+8 = 1288 halves (2576B: 16B aligned, odd word stride/4).
#define ROWP    40
#define SROW    (GDIM * ROWP + 8)      // 1288 halves
#define TILE_H  (SB * SROW)            // 41216 halves = 82432 B

// f32 staging for transposed output stores: [g][f][s], f stride 33, g stride 1057
#define FSTR    33
#define GSTR    (32 * FSTR + 1)        // 1057
#define STAGE_F (32 * GSTR)            // 33824 floats = 135296 B

#define SMEM_BYTES (2 * TILE_H * 2 + STAGE_F * 4)   // 300160 B dynamic LDS

__device__ __forceinline__ v16h ld_frag(const _Float16* p, int off0, int off1) {
  union { v16h v; h8 h[2]; } u;
  u.h[0] = *(const h8*)(p + off0);   // ds_load_b128
  u.h[1] = *(const h8*)(p + off1);   // ds_load_b128
  return u.v;
}

__global__ void __launch_bounds__(1024)
tamcad_attn_kernel(const float* __restrict__ q,
                   const float* __restrict__ k,
                   const float* __restrict__ v,
                   float* __restrict__ x_out,
                   float* __restrict__ attn_out,
                   float* __restrict__ logit_out) {
  extern __shared__ char smem[];
  _Float16* qtile = (_Float16*)smem;            // Q [g][d], later attn [g][f]
  _Float16* ktile = qtile + TILE_H;             // K [f][d], later V^T [d][f]
  float*    stage = (float*)(smem + (size_t)2 * TILE_H * 2);

  const int lane = threadIdx.x & 31;
  const int wv   = threadIdx.x >> 5;            // wave id == local s index
  const int bb   = blockIdx.y;
  const int s0   = blockIdx.x * SB;

  const size_t gbase = ((size_t)bb * GDIM) * DDIM * S_TOTAL + (size_t)s0 + lane;

  // ---- Stage Q [g][d] and K [f][d] (f32 global -> f16 LDS), coalesced ----
  #pragma unroll 4
  for (int r = 0; r < GDIM; ++r) {              // r = g (Q) / f (K); col = wv = d
    size_t gi = gbase + ((size_t)r * DDIM + wv) * S_TOTAL;
    int li = lane * SROW + r * ROWP + wv;
    qtile[li] = (_Float16)q[gi];
    ktile[li] = (_Float16)k[gi];
  }
  __syncthreads();

  // ---- logits = scale * Q_s K_s^T : 4x v_wmma_f32_16x16x32_f16 ----
  v8f acc[4];                                   // [ti*2+tj], C/D layout
  {
    const _Float16* qs = qtile + wv * SROW;
    const _Float16* ks = ktile + wv * SROW;
    const int lm  = lane & 15;
    const int akb = (lane < 16) ? 0 : 8;        // A: lanes 0-15 K=0-7/16-23, hi: 8-15/24-31
    const int bkb = (lane < 16) ? 0 : 8;        // B: lanes 0-15 K=0-7/16-23, hi: 8-15/24-31
    v16h A[2], Bm[2];
    #pragma unroll
    for (int t = 0; t < 2; ++t) {
      int m = t * 16 + lm;                      // A row = g ; B col n = f (same index expr)
      A[t]  = ld_frag(qs, m * ROWP + akb, m * ROWP + 16 + akb);
      Bm[t] = ld_frag(ks, m * ROWP + bkb, m * ROWP + 16 + bkb);
    }
    #pragma unroll
    for (int ti = 0; ti < 2; ++ti)
      #pragma unroll
      for (int tj = 0; tj < 2; ++tj) {
        v8f c = {0.f, 0.f, 0.f, 0.f, 0.f, 0.f, 0.f, 0.f};
        acc[ti * 2 + tj] = __builtin_amdgcn_wmma_f32_16x16x32_f16(
            false, A[ti], false, Bm[tj], (short)0, c, false, false);
      }
    const float scale = 0.17677669529663687f;   // 32^-0.5
    #pragma unroll
    for (int t = 0; t < 4; ++t)
      #pragma unroll
      for (int j = 0; j < 8; ++j)
        acc[t][j] *= scale;
  }

  const int rhalf = (lane >> 4) * 8;            // C/D: vgpr j -> row j (+8 for hi lanes)
  const int cn    = lane & 15;

  // ---- logits -> stage -> coalesced global store ----
  #pragma unroll
  for (int ti = 0; ti < 2; ++ti)
    #pragma unroll
    for (int tj = 0; tj < 2; ++tj) {
      v8f a = acc[ti * 2 + tj];
      #pragma unroll
      for (int j = 0; j < 8; ++j) {
        int g = ti * 16 + j + rhalf;
        int f = tj * 16 + cn;
        stage[g * GSTR + f * FSTR + wv] = a[j];
      }
    }
  __syncthreads();
  #pragma unroll 4
  for (int i = 0; i < 32; ++i) {
    int row = i * 32 + wv;                      // row = g*32 + f
    int g = row >> 5, f = row & 31;
    logit_out[((size_t)bb * 1024 + row) * S_TOTAL + s0 + lane] =
        stage[g * GSTR + f * FSTR + lane];
  }
  __syncthreads();                              // stage reusable

  // ---- Stage V transposed: [d][f] (K tile is dead) ----
  #pragma unroll 4
  for (int d = 0; d < GDIM; ++d) {              // f = wv, rows indexed by d
    float vv = v[gbase + ((size_t)wv * DDIM + d) * S_TOTAL];
    ktile[lane * SROW + d * ROWP + wv] = (_Float16)vv;
  }

  // ---- softmax over f (registers + 16-lane butterflies) ----
  #pragma unroll
  for (int ti = 0; ti < 2; ++ti) {
    #pragma unroll
    for (int j = 0; j < 8; ++j) {
      float v0 = acc[ti * 2 + 0][j];
      float v1 = acc[ti * 2 + 1][j];
      float mx = fmaxf(v0, v1);
      #pragma unroll
      for (int off = 1; off < 16; off <<= 1)
        mx = fmaxf(mx, __shfl_xor(mx, off, 32));
      float e0 = __expf(v0 - mx);
      float e1 = __expf(v1 - mx);
      float sm = e0 + e1;
      #pragma unroll
      for (int off = 1; off < 16; off <<= 1)
        sm += __shfl_xor(sm, off, 32);
      float inv = __builtin_amdgcn_rcpf(sm);
      acc[ti * 2 + 0][j] = e0 * inv;
      acc[ti * 2 + 1][j] = e1 * inv;
    }
  }

  // ---- attn -> stage (f32, for store) and -> qtile (f16 A-fragment [g][f]) ----
  #pragma unroll
  for (int ti = 0; ti < 2; ++ti)
    #pragma unroll
    for (int tj = 0; tj < 2; ++tj) {
      v8f a = acc[ti * 2 + tj];
      #pragma unroll
      for (int j = 0; j < 8; ++j) {
        int g = ti * 16 + j + rhalf;
        int f = tj * 16 + cn;
        stage[g * GSTR + f * FSTR + wv] = a[j];
        qtile[wv * SROW + g * ROWP + f] = (_Float16)a[j];
      }
    }
  __syncthreads();                              // attn staged + V staged everywhere
  #pragma unroll 4
  for (int i = 0; i < 32; ++i) {
    int row = i * 32 + wv;
    int g = row >> 5, f = row & 31;
    attn_out[((size_t)bb * 1024 + row) * S_TOTAL + s0 + lane] =
        stage[g * GSTR + f * FSTR + lane];
  }

  // ---- X = attn * V : 4x v_wmma_f32_16x16x32_f16 ----
  v8f xacc[4];
  {
    const _Float16* as = qtile + wv * SROW;     // attn [g][f]
    const _Float16* vs = ktile + wv * SROW;     // V^T  [d][f]
    const int lm  = lane & 15;
    const int akb = (lane < 16) ? 0 : 8;
    const int bkb = (lane < 16) ? 0 : 8;
    v16h A[2], Bm[2];
    #pragma unroll
    for (int t = 0; t < 2; ++t) {
      int m = t * 16 + lm;                      // A row = g ; B col n = d
      A[t]  = ld_frag(as, m * ROWP + akb, m * ROWP + 16 + akb);
      Bm[t] = ld_frag(vs, m * ROWP + bkb, m * ROWP + 16 + bkb);
    }
    #pragma unroll
    for (int ti = 0; ti < 2; ++ti)
      #pragma unroll
      for (int tj = 0; tj < 2; ++tj) {
        v8f c = {0.f, 0.f, 0.f, 0.f, 0.f, 0.f, 0.f, 0.f};
        xacc[ti * 2 + tj] = __builtin_amdgcn_wmma_f32_16x16x32_f16(
            false, A[ti], false, Bm[tj], (short)0, c, false, false);
      }
  }
  __syncthreads();                              // stage free (attn store done)

  // ---- X -> stage -> coalesced global store ----
  #pragma unroll
  for (int ti = 0; ti < 2; ++ti)
    #pragma unroll
    for (int tj = 0; tj < 2; ++tj) {
      v8f a = xacc[ti * 2 + tj];
      #pragma unroll
      for (int j = 0; j < 8; ++j) {
        int g  = ti * 16 + j + rhalf;
        int dc = tj * 16 + cn;
        stage[g * GSTR + dc * FSTR + wv] = a[j];
      }
    }
  __syncthreads();
  #pragma unroll 4
  for (int i = 0; i < 32; ++i) {
    int row = i * 32 + wv;                      // row = g*32 + d
    int g = row >> 5, dd = row & 31;
    x_out[((size_t)bb * 1024 + row) * S_TOTAL + s0 + lane] =
        stage[g * GSTR + dd * FSTR + lane];
  }
}

extern "C" void kernel_launch(void* const* d_in, const int* in_sizes, int n_in,
                              void* d_out, int out_size, void* d_ws, size_t ws_size,
                              hipStream_t stream) {
  (void)in_sizes; (void)n_in; (void)out_size; (void)d_ws; (void)ws_size;
  const float* q = (const float*)d_in[0];
  const float* k = (const float*)d_in[1];
  const float* v = (const float*)d_in[2];
  float* out = (float*)d_out;
  const size_t one = (size_t)16 * 1024 * 4096;  // elements per output tensor
  float* x_out     = out;
  float* attn_out  = out + one;
  float* logit_out = out + 2 * one;

  static_assert(SMEM_BYTES == 300160, "LDS budget");
  (void)hipFuncSetAttribute((const void*)tamcad_attn_kernel,
                            hipFuncAttributeMaxDynamicSharedMemorySize,
                            SMEM_BYTES);

  dim3 grid(S_TOTAL / SB, 16);                  // (128 s-blocks, 16 batches)
  tamcad_attn_kernel<<<grid, 1024, SMEM_BYTES, stream>>>(q, k, v,
                                                         x_out, attn_out, logit_out);
}
